// GraphUNetSmall_43018392436874
// MI455X (gfx1250) — matrix operand order repaired and checked by reading, not compiled, not run on hardware
//
#include <hip/hip_runtime.h>

// ---------------------------------------------------------------------------
// GraphUNet on MI455X (gfx1250).
// gconv(x,ei,ea,W,b,n) == relu(scatter_w(x,ei,ea,n) @ W + b)  (W commutes out
// of the edge loop). Dual-dtype pipeline:
//   - scatter destinations: f32 (global_atomic_add_f32, exact accumulation;
//     every destination buffer <=64MB -> L2-resident at 192MB)
//   - activations (GEMM / concat outputs): bf16 in memory. The WMMA path
//     rounds A to bf16 anyway, so this halves all edge-gather bytes for free.
// Node GEMMs: v_wmma_f32_16x16x32_bf16, 16-row tile per wave32, weights+bias
// staged in registers once per wave, grid-stride over tiles. All bf16<->f32
// conversions via native hardware converts (vector convertvector).
// ---------------------------------------------------------------------------

typedef __attribute__((ext_vector_type(16))) __bf16        v16bf;
typedef __attribute__((ext_vector_type(8)))  __bf16        bf8v;
typedef __attribute__((ext_vector_type(4)))  __bf16        bf4v;
typedef __attribute__((ext_vector_type(8)))  float         v8f;
typedef __attribute__((ext_vector_type(4)))  float         f4v;
typedef __attribute__((ext_vector_type(2)))  float         f2v;
typedef __attribute__((ext_vector_type(4)))  unsigned int  u4v;   // 8 packed bf16

__device__ __forceinline__ unsigned short f2bf(float f) {
  union { __bf16 b; unsigned short u; } cv;
  cv.b = (__bf16)f;                  // native cvt, round-to-nearest-even
  return cv.u;
}
__device__ __forceinline__ float ldf(const float* p) { return *p; }
__device__ __forceinline__ float ldf(const unsigned short* p) {
  return __uint_as_float((unsigned int)(*p) << 16);   // bf16 -> f32 is exact
}

// n4 = count of float4s (all scatter-dest sizes are multiples of 4 floats)
__global__ __launch_bounds__(256) void zerof_kernel(f4v* __restrict__ p, long long n4) {
  long long i = (long long)blockIdx.x * blockDim.x + threadIdx.x;
  if (i < n4) p[i] = (f4v){0.f, 0.f, 0.f, 0.f};
}

// f32-source weighted scatter-add: out[dst] += ea * x[src]
template<int F>
__global__ __launch_bounds__(256) void scatter_f32_kernel(
    const float* __restrict__ x, const long long* __restrict__ ei,
    const float* __restrict__ ea, float* __restrict__ out, long long E) {
  long long e = (long long)blockIdx.x * blockDim.x + threadIdx.x;
  if (e >= E) return;
  long long s = ei[e], d = ei[E + e];
  float w = ea[e];
  const float* xs = x + s * F;
  float* od = out + d * F;
  if constexpr (F == 2) {
    f2v v = *(const f2v*)xs;
    atomicAdd(od + 0, w * v.x);
    atomicAdd(od + 1, w * v.y);
  } else {
#pragma unroll
    for (int q = 0; q < F / 4; ++q) {
      f4v v = *(const f4v*)(xs + 4 * q);
#pragma unroll
      for (int r = 0; r < 4; ++r) atomicAdd(od + 4 * q + r, w * v[r]);
    }
  }
}

// bf16-source weighted scatter-add (half the gather bytes): b128 = 8 bf16
template<int F>
__global__ __launch_bounds__(256) void scatter_bf16_kernel(
    const unsigned short* __restrict__ x, const long long* __restrict__ ei,
    const float* __restrict__ ea, float* __restrict__ out, long long E) {
  long long e = (long long)blockIdx.x * blockDim.x + threadIdx.x;
  if (e >= E) return;
  long long s = ei[e], d = ei[E + e];
  float w = ea[e];
  const unsigned int* xs = (const unsigned int*)(x + s * F);
  float* od = out + d * F;
#pragma unroll
  for (int q = 0; q < F / 8; ++q) {
    union { u4v u; bf8v b; } c;
    c.u = *(const u4v*)(xs + 4 * q);
    v8f f = __builtin_convertvector(c.b, v8f);   // packed bf16->f32
#pragma unroll
    for (int r = 0; r < 8; ++r) atomicAdd(od + 8 * q + r, w * f[r]);
  }
}

// concat along features; mixed input dtypes, bf16 output
template<int Fa, int Fb, typename TA, typename TB>
__global__ __launch_bounds__(256) void concat2_kernel(
    const TA* __restrict__ a, const TB* __restrict__ b,
    unsigned short* __restrict__ o, long long n) {
  constexpr int F = Fa + Fb;
  long long i = (long long)blockIdx.x * blockDim.x + threadIdx.x;
  if (i >= n * F) return;
  long long node = i / F;             // division by constant -> mul/shift
  int f = (int)(i - node * F);
  float v = (f < Fa) ? ldf(a + node * Fa + f) : ldf(b + node * Fb + (f - Fa));
  o[i] = f2bf(v);
}

// Y[n,FO](bf16) = relu(X[n,FI](f32) @ W[FI,FO] + bias).
// Fragment layouts per CDNA5 ISA 7.12.2:
//  A 16x32 bf16: lane row=lane%16; elems 0..7 -> K k0..k0+7, 8..15 -> k0+16..k0+23,
//                k0 = 0 (lanes 0-15) / 8 (lanes 16-31)  -> float4-group loads.
//  B 32x16 bf16: col=lane%16; elems 0..7 -> K k0..k0+7, 8..15 -> k0+16..k0+23 (same k0 as A).
//  C/D 16x16 f32: VGPR j -> row j / j+8 per lane half, col=lane%16.
template<int FI, int FO>
__global__ __launch_bounds__(256) void node_gemm_relu_kernel(
    const float* __restrict__ X, const float* __restrict__ W,
    const float* __restrict__ bias, unsigned short* __restrict__ Y, long long n) {
  constexpr int KT = (FI + 31) / 32;
  constexpr int NT = (FO + 15) / 16;
  const int lane = threadIdx.x & 31;
  const bool hi = lane >= 16;
  const int l16 = lane & 15;
  const int k0off = hi ? 8 : 0;

  union Frag { v16bf v; bf4v q[4]; unsigned short u[16]; };

  // ---- stage B fragments + bias in registers once per wave ----
  Frag bm[KT][NT];
#pragma unroll
  for (int kt = 0; kt < KT; ++kt)
#pragma unroll
    for (int t = 0; t < NT; ++t) {
      int col = t * 16 + l16;
      f4v g[4];
#pragma unroll
      for (int j = 0; j < 16; ++j) {
        int k = kt * 32 + (j < 8 ? j : j + 8) + (hi ? 8 : 0);
        g[j >> 2][j & 3] = (col < FO && k < FI) ? W[(long long)k * FO + col] : 0.0f;
      }
#pragma unroll
      for (int q = 0; q < 4; ++q)
        bm[kt][t].q[q] = __builtin_convertvector(g[q], bf4v);
    }
  float bv[NT];
#pragma unroll
  for (int t = 0; t < NT; ++t) {
    int col = t * 16 + l16;
    bv[t] = (col < FO) ? bias[col] : 0.0f;
  }

  // ---- grid-stride over 16-row tiles ----
  const long long tiles = (n + 15) >> 4;
  const long long tstep = (long long)gridDim.x * (blockDim.x >> 5);
  for (long long tile = (long long)blockIdx.x * (blockDim.x >> 5) + (threadIdx.x >> 5);
       tile < tiles; tile += tstep) {
    const long long rowBase = tile * 16;
    const long long gr = rowBase + l16;        // A row for this lane
    const bool rowOk = gr < n;
    const float* xr = X + gr * (long long)FI;

    v8f acc[NT];
#pragma unroll
    for (int t = 0; t < NT; ++t) acc[t] = (v8f){0.f,0.f,0.f,0.f,0.f,0.f,0.f,0.f};

#pragma unroll
    for (int kt = 0; kt < KT; ++kt) {
      Frag a;
      if constexpr (FI == 2) {
        // only K=0,1 exist: lanes 0-15 get them, lanes 16-31 (k0=8) are zero
        f2v v = (rowOk && !hi) ? *(const f2v*)xr : (f2v){0.f, 0.f};
        a.u[0] = f2bf(v.x); a.u[1] = f2bf(v.y);
#pragma unroll
        for (int j = 2; j < 16; ++j) a.u[j] = 0;
      } else {
        // FI % 8 == 0: each 8-element group (K kg..kg+7) is all-valid or all-zero
        const int kg0 = kt * 32 + k0off;
        const int kg1 = kg0 + 16;
        const f4v z4 = {0.f, 0.f, 0.f, 0.f};
        f4v va0 = (rowOk && kg0 < FI) ? *(const f4v*)(xr + kg0)     : z4;
        f4v va1 = (rowOk && kg0 < FI) ? *(const f4v*)(xr + kg0 + 4) : z4;
        f4v vb0 = (rowOk && kg1 < FI) ? *(const f4v*)(xr + kg1)     : z4;
        f4v vb1 = (rowOk && kg1 < FI) ? *(const f4v*)(xr + kg1 + 4) : z4;
        a.q[0] = __builtin_convertvector(va0, bf4v);   // packed f32->bf16
        a.q[1] = __builtin_convertvector(va1, bf4v);
        a.q[2] = __builtin_convertvector(vb0, bf4v);
        a.q[3] = __builtin_convertvector(vb1, bf4v);
      }
#pragma unroll
      for (int t = 0; t < NT; ++t)
        acc[t] = __builtin_amdgcn_wmma_f32_16x16x32_bf16(
            false, a.v, false, bm[kt][t].v, (short)0, acc[t], false, false);
    }

    // ---- epilogue: scalar (wave-uniform) full-tile branch; one lane mask per
    //      n-tile (compile-time true when FO%16==0); packed cvt + bf16 stores
    const bool full = __builtin_amdgcn_readfirstlane((int)(rowBase + 16 <= n)) != 0;
    const int mOff = hi ? 8 : 0;
#pragma unroll
    for (int t = 0; t < NT; ++t) {
      int col = t * 16 + l16;
      if (col < FO) {
        f4v lo4, hi4;
#pragma unroll
        for (int j = 0; j < 4; ++j) {
          float a0 = acc[t][j] + bv[t];
          float a1 = acc[t][j + 4] + bv[t];
          lo4[j] = a0 > 0.0f ? a0 : 0.0f;
          hi4[j] = a1 > 0.0f ? a1 : 0.0f;
        }
        union { bf4v b; unsigned short u[4]; } e0, e1;
        e0.b = __builtin_convertvector(lo4, bf4v);
        e1.b = __builtin_convertvector(hi4, bf4v);
        unsigned short* yc = Y + (rowBase + mOff) * (long long)FO + col;
        if (full) {
#pragma unroll
          for (int j = 0; j < 4; ++j) {
            yc[(long long)j * FO]       = e0.u[j];
            yc[(long long)(j + 4) * FO] = e1.u[j];
          }
        } else {
#pragma unroll
          for (int j = 0; j < 8; ++j) {
            if (rowBase + mOff + j < n)
              yc[(long long)j * FO] = (j < 4) ? e0.u[j] : e1.u[j - 4];
          }
        }
      }
    }
  }
}

// out[i] = X[i,0:16](bf16) . Wf + bf   (final readout, FO=1 -> plain VALU)
__global__ __launch_bounds__(256) void readout_kernel(
    const unsigned short* __restrict__ X, const float* __restrict__ Wf,
    const float* __restrict__ bf_, float* __restrict__ out, long long n) {
  long long i = (long long)blockIdx.x * blockDim.x + threadIdx.x;
  if (i >= n) return;
  const u4v* xr = (const u4v*)(X + i * 16);   // 16 bf16 = 2 x b128
  float s = bf_[0];
#pragma unroll
  for (int q = 0; q < 2; ++q) {
    union { u4v u; bf8v b; } c;
    c.u = xr[q];
    v8f f = __builtin_convertvector(c.b, v8f);
#pragma unroll
    for (int r = 0; r < 8; ++r) s += f[r] * Wf[8 * q + r];
  }
  out[i] = s;
}

// ---------------------------------------------------------------------------
// Host-side helpers
// ---------------------------------------------------------------------------
static inline int gridFor(long long n, int b) { return (int)((n + b - 1) / b); }

static void zero_dest(hipStream_t st, float* out, long long nfloats) {
  long long n4 = nfloats / 4;        // all dest sizes are multiples of 4 floats
  zerof_kernel<<<gridFor(n4, 256), 256, 0, st>>>((f4v*)out, n4);
}

static void scatw_f32(hipStream_t st, int F, const float* x, const void* ei_,
                      const float* ea, long long E, float* out, long long nout) {
  const long long* ei = (const long long*)ei_;
  zero_dest(st, out, nout * F);
  int g = gridFor(E, 256);
  switch (F) {
    case 2:  scatter_f32_kernel<2> <<<g, 256, 0, st>>>(x, ei, ea, out, E); break;
    case 16: scatter_f32_kernel<16><<<g, 256, 0, st>>>(x, ei, ea, out, E); break;
    case 32: scatter_f32_kernel<32><<<g, 256, 0, st>>>(x, ei, ea, out, E); break;
  }
}

static void scatw_bf16(hipStream_t st, int F, const unsigned short* x, const void* ei_,
                       const float* ea, long long E, float* out, long long nout) {
  const long long* ei = (const long long*)ei_;
  zero_dest(st, out, nout * F);
  int g = gridFor(E, 256);
  switch (F) {
    case 16: scatter_bf16_kernel<16><<<g, 256, 0, st>>>(x, ei, ea, out, E); break;
    case 32: scatter_bf16_kernel<32><<<g, 256, 0, st>>>(x, ei, ea, out, E); break;
    case 48: scatter_bf16_kernel<48><<<g, 256, 0, st>>>(x, ei, ea, out, E); break;
    case 64: scatter_bf16_kernel<64><<<g, 256, 0, st>>>(x, ei, ea, out, E); break;
  }
}

static void gemm(hipStream_t st, int FI, int FO, const float* X, const float* W,
                 const float* b, unsigned short* Y, long long n) {
  long long tiles = (n + 15) / 16;
  int g = gridFor(tiles, 8);          // 8 wave-tiles per 256-thread block
  if (g > 3072) g = 3072;             // grid-stride amortizes B/bias staging
#define NG_CASE(fi, fo) \
  if (FI == fi && FO == fo) { node_gemm_relu_kernel<fi, fo><<<g, 256, 0, st>>>(X, W, b, Y, n); return; }
  NG_CASE(2, 12) NG_CASE(16, 16) NG_CASE(16, 32) NG_CASE(32, 32)
  NG_CASE(64, 32) NG_CASE(48, 32) NG_CASE(48, 16) NG_CASE(32, 16)
#undef NG_CASE
}

template<int Fa, int Fb, typename TA, typename TB>
static void cat2_t(hipStream_t st, const TA* a, const TB* b, unsigned short* o, long long n) {
  long long tot = n * (long long)(Fa + Fb);
  concat2_kernel<Fa, Fb, TA, TB><<<gridFor(tot, 256), 256, 0, st>>>(a, b, o, n);
}

extern "C" void kernel_launch(void* const* d_in, const int* in_sizes, int n_in,
                              void* d_out, int out_size, void* d_ws, size_t ws_size,
                              hipStream_t stream) {
  const long long NCn = 1000000, NFn = 1500000, N0n = 500000, N1n = 125000,
                  N2n = 31250, N3n = 7813, N4n = 1954;
  typedef unsigned short bf_t;

  const float* xc = (const float*)d_in[0];
  const float* xf = (const float*)d_in[1];

  // edge relations: d_in[i] = ei (int64 [2,E]), d_in[i+1] = ea (f32 [E])
  const int I_CF = 2, I_FP = 4, I_PC = 6, I_PP0 = 8, I_PP1 = 10, I_PP2 = 12,
            I_PP3 = 14, I_PP4 = 16, I_PL0 = 18, I_PL1 = 20, I_PL2 = 22,
            I_PL3 = 24, I_UP0 = 26, I_UP1 = 28, I_UP2 = 30, I_UP3 = 32;
  auto EA = [&](int i) { return (const float*)d_in[i + 1]; };
  auto EC = [&](int i) { return (long long)in_sizes[i] / 2; };
  auto WP = [&](int i) { return (const float*)d_in[i]; };

  const float *W_cf = WP(34), *b_cf = WP(35), *W_fp = WP(36), *b_fp = WP(37),
              *W2 = WP(38),  *b2 = WP(39),  *W3 = WP(40),  *b3 = WP(41),
              *W4 = WP(42),  *b4 = WP(43),  *W5a = WP(44), *b5a = WP(45),
              *W5b = WP(46), *b5b = WP(47), *W6 = WP(48),  *b6 = WP(49),
              *W7 = WP(50),  *b7 = WP(51),  *W8 = WP(52),  *b8 = WP(53),
              *W9a = WP(54), *b9a = WP(55), *W9b = WP(56), *b9b = WP(57),
              *Wf = WP(58),  *bf_ = WP(59);

  // Workspace (byte offsets, explicit liveness reuse; peak ~166 MB).
  char* wsb = (char*)d_ws;
  auto F32 = [&](long long off) { return (float*)(wsb + off); };
  auto BF  = [&](long long off) { return (bf_t*)(wsb + off); };

  bf_t* c1 = BF(0);                    // [N0,16] bf16, live to end
  bf_t* c2 = BF(16000000LL);           // [N1,16]
  bf_t* c3 = BF(20000000LL);           // [N2,16]
  bf_t* c4 = BF(21000000LL);           // [N3,32]
  const long long AB = 22000000LL;     // arena base

  // ---- Phase 1: centers -> faces -> points ----
  float* hpre  = F32(AB);              // [NF,2]  f32 (12MB)
  bf_t*  h     = BF(AB + 48000000LL);  // [NF,12] bf16 (36MB)
  scatw_f32(stream, 2, xc, d_in[I_CF], EA(I_CF), EC(I_CF), hpre, NFn);
  gemm(stream, 2, 12, hpre, W_cf, b_cf, h, NFn);
  bf_t*  hx    = BF(AB);               // [NF,16] bf16 (48MB, overwrites hpre)
  cat2_t<12, 4>(stream, h, xf, hx, NFn);
  float* c1pre = F32(AB + 48000000LL); // [N0,16] f32 (32MB, overwrites h)
  scatw_bf16(stream, 16, hx, d_in[I_FP], EA(I_FP), EC(I_FP), c1pre, N0n);
  gemm(stream, 16, 16, c1pre, W_fp, b_fp, c1, N0n);

  // ---- Phase 2: encoder ----
  float* s = F32(AB);                  // [N1,16] f32
  float* t = F32(AB + 8000000LL);
  scatw_bf16(stream, 16, c1, d_in[I_PL0], EA(I_PL0), EC(I_PL0), s, N1n);
  scatw_f32 (stream, 16, s,  d_in[I_PP1], EA(I_PP1), EC(I_PP1), t, N1n);
  gemm(stream, 16, 16, t, W2, b2, c2, N1n);
  scatw_bf16(stream, 16, c2, d_in[I_PL1], EA(I_PL1), EC(I_PL1), s, N2n);
  scatw_f32 (stream, 16, s,  d_in[I_PP2], EA(I_PP2), EC(I_PP2), t, N2n);
  gemm(stream, 16, 16, t, W3, b3, c3, N2n);
  scatw_bf16(stream, 16, c3, d_in[I_PL2], EA(I_PL2), EC(I_PL2), s, N3n);
  scatw_f32 (stream, 16, s,  d_in[I_PP3], EA(I_PP3), EC(I_PP3), t, N3n);
  gemm(stream, 16, 32, t, W4, b4, c4, N3n);

  // ---- Phase 3: bottleneck + decoder ----
  float* p4    = F32(AB);               // [N4,32] f32
  float* q4    = F32(AB + 400000LL);
  bf_t*  c5a   = BF (AB + 800000LL);    // [N4,32] bf16
  float* q4b   = F32(AB + 1000000LL);
  bf_t*  c5b   = BF (AB + 1300000LL);
  float* c5u   = F32(AB + 1500000LL);   // [N3,32] f32
  bf_t*  cat54 = BF (AB + 2600000LL);   // [N3,64] bf16
  float* u6    = F32(AB + 3700000LL);   // [N3,64] f32
  bf_t*  w6o   = BF (AB + 5800000LL);   // [N3,32] bf16
  float* c6    = F32(AB + 6400000LL);   // [N2,32] f32
  bf_t*  cat63 = BF (AB + 10400000LL);  // [N2,48] bf16
  float* v7    = F32(AB + 13400000LL);  // [N2,48] f32
  bf_t*  w7o   = BF (AB + 19400000LL);  // [N2,32] bf16
  float* c7    = F32(AB + 21400000LL);  // [N1,32] f32
  bf_t*  cat72 = BF (AB + 37400000LL);  // [N1,48] bf16
  float* v8    = F32(AB + 49400000LL);  // [N1,48] f32
  bf_t*  w8o   = BF (AB + 73400000LL);  // [N1,16] bf16
  float* c8    = F32(AB + 77400000LL);  // [N0,16] f32
  bf_t*  cat81 = BF (AB + 109400000LL); // [N0,32] bf16 (peak)
  float* pre9  = F32(AB);               // [N0,32] f32  reuse (p4..v8 dead)
  bf_t*  c9    = BF (AB + 64000000LL);  // [N0,16] bf16
  float* c9p   = F32(AB + 80000000LL);  // [NC,16] f32  (c8/cat81 dead)
  bf_t*  c9b   = BF (AB);               // [NC,16] bf16 reuse (pre9 dead)

  scatw_bf16(stream, 32, c4,  d_in[I_PL3], EA(I_PL3), EC(I_PL3), p4,  N4n);
  scatw_f32 (stream, 32, p4,  d_in[I_PP4], EA(I_PP4), EC(I_PP4), q4,  N4n);
  gemm(stream, 32, 32, q4, W5a, b5a, c5a, N4n);
  scatw_bf16(stream, 32, c5a, d_in[I_PP4], EA(I_PP4), EC(I_PP4), q4b, N4n);
  gemm(stream, 32, 32, q4b, W5b, b5b, c5b, N4n);
  scatw_bf16(stream, 32, c5b, d_in[I_UP3], EA(I_UP3), EC(I_UP3), c5u, N3n);

  cat2_t<32, 32>(stream, c5u, c4, cat54, N3n);
  scatw_bf16(stream, 64, cat54, d_in[I_PP3], EA(I_PP3), EC(I_PP3), u6, N3n);
  gemm(stream, 64, 32, u6, W6, b6, w6o, N3n);
  scatw_bf16(stream, 32, w6o, d_in[I_UP2], EA(I_UP2), EC(I_UP2), c6, N2n);

  cat2_t<32, 16>(stream, c6, c3, cat63, N2n);
  scatw_bf16(stream, 48, cat63, d_in[I_PP2], EA(I_PP2), EC(I_PP2), v7, N2n);
  gemm(stream, 48, 32, v7, W7, b7, w7o, N2n);
  scatw_bf16(stream, 32, w7o, d_in[I_UP1], EA(I_UP1), EC(I_UP1), c7, N1n);

  cat2_t<32, 16>(stream, c7, c2, cat72, N1n);
  scatw_bf16(stream, 48, cat72, d_in[I_PP1], EA(I_PP1), EC(I_PP1), v8, N1n);
  gemm(stream, 48, 16, v8, W8, b8, w8o, N1n);
  scatw_bf16(stream, 16, w8o, d_in[I_UP0], EA(I_UP0), EC(I_UP0), c8, N0n);

  cat2_t<16, 16>(stream, c8, c1, cat81, N0n);
  scatw_bf16(stream, 32, cat81, d_in[I_PP0], EA(I_PP0), EC(I_PP0), pre9, N0n);
  gemm(stream, 32, 16, pre9, W9a, b9a, c9, N0n);
  scatw_bf16(stream, 16, c9, d_in[I_PC], EA(I_PC), EC(I_PC), c9p, NCn);
  gemm(stream, 16, 16, c9p, W9b, b9b, c9b, NCn);

  readout_kernel<<<gridFor(NCn, 256), 256, 0, stream>>>(c9b, Wf, bf_, (float*)d_out, NCn);

  (void)n_in; (void)out_size; (void)ws_size;
}
